// Seq2SeqLSTM_67413806678319
// MI455X (gfx1250) — matrix operand order repaired and checked by reading, not compiled, not run on hardware
//
#include <hip/hip_runtime.h>
#include <hip/hip_bf16.h>

// Problem constants (match reference)
#define BATCH  1024
#define SEQT   512
#define FEAT   128
#define HID    128
#define GATES  512      // 4*H
#define OUTLEN 15

typedef __attribute__((ext_vector_type(16))) __bf16 v16bf;
typedef __attribute__((ext_vector_type(8)))  float  v8f;

// gfx1250 async global->LDS path (ASYNCcnt-tracked), with safe fallback.
#if defined(__gfx1250__) && __has_builtin(__builtin_amdgcn_global_load_async_to_lds_b128)
#define HAVE_ASYNC 1
#else
#define HAVE_ASYNC 0
#endif

#if HAVE_ASYNC
// Exact parameter types per hipcc diagnostic:
//   param0: int __attribute__((vector_size(16))) __device__*   (global src)
//   param1: LDS destination pointer
typedef int async_v4i __attribute__((vector_size(4 * sizeof(int))));
typedef __attribute__((address_space(1))) async_v4i* async_gp;
typedef __attribute__((address_space(3))) async_v4i* async_sp;

__device__ __forceinline__ void async_copy_b128(const void* gsrc, void* ldst) {
  __builtin_amdgcn_global_load_async_to_lds_b128(
      (async_gp)(void*)gsrc, (async_sp)ldst, 0, 0);
}

__device__ __forceinline__ void wait_async_le1() {
#if __has_builtin(__builtin_amdgcn_s_wait_asynccnt)
  __builtin_amdgcn_s_wait_asynccnt(1);
#else
  asm volatile("s_wait_asynccnt 0x1" ::: "memory");
#endif
}
#endif

__device__ __forceinline__ v8f zero8() {
  v8f z;
  #pragma unroll
  for (int i = 0; i < 8; ++i) z[i] = 0.0f;
  return z;
}

__device__ __forceinline__ float sigmoidf(float x) {
  return 1.0f / (1.0f + __expf(-x));
}

// ---------------------------------------------------------------------------
// WMMA fragment helpers for V_WMMA_F32_16X16X32_BF16 (wave32).
// A: 16x32 bf16. Lanes 0-15 hold M=lane, K=0..15 sequentially;
// lanes 16-31 hold K=16..31 (matches B fragment order).
// ---------------------------------------------------------------------------
__device__ __forceinline__ v16bf load_A_frag(const __bf16* tile /*[16][128]*/,
                                             int lane, int kk /*k-step*/) {
  const int m    = lane & 15;
  const int half = lane >> 4;
  const __bf16* row = tile + m * 128 + kk * 32;
  v16bf a;
  #pragma unroll
  for (int j = 0; j < 8; ++j) {
    const int kb = 2 * j + half * 16;
    a[2 * j]     = row[kb];
    a[2 * j + 1] = row[kb + 1];
  }
  return a;
}

// B: 32x16 bf16, B[k][n] = W[n][k] for out = in @ W.T (W row-major [N][ldk]).
// Lanes 0-15 hold K=0..15 sequentially, lanes 16-31 hold K=16..31; col=lane&15.
__device__ __forceinline__ v16bf load_B_frag_lds(const __bf16* w, int lane,
                                                 int n_base, int kk, int ldk) {
  const int n    = n_base + (lane & 15);
  const int half = lane >> 4;
  const __bf16* row = w + n * ldk + kk * 32 + half * 16;
  v16bf b;
  #pragma unroll
  for (int j = 0; j < 16; ++j) b[j] = row[j];
  return b;
}

__device__ __forceinline__ v16bf load_B_frag_gf32(const float* w, int lane,
                                                  int n_base, int kk, int ldk) {
  const int n    = n_base + (lane & 15);
  const int half = lane >> 4;
  const float* row = w + n * ldk + kk * 32 + half * 16;
  v16bf b;
  #pragma unroll
  for (int j = 0; j < 16; ++j) b[j] = (__bf16)row[j];
  return b;
}

// D/C: 8 f32 per lane; VGPR r holds row (r + 8*half), col = lane&15.
__device__ __forceinline__ void store_D(float* out, int lane, int n_base,
                                        int ldn, v8f acc) {
  const int n    = n_base + (lane & 15);
  const int half = lane >> 4;
  #pragma unroll
  for (int r = 0; r < 8; ++r) out[(r + 8 * half) * ldn + n] = acc[r];
}

#define WMMA_BF16(a, b, c) \
  __builtin_amdgcn_wmma_f32_16x16x32_bf16(false, (a), false, (b), (short)0, (c), false, false)

// ---------------------------------------------------------------------------
// f32 -> bf16 cast of x (once), so both encoder layers use the 16B-aligned
// async-to-LDS staging path.
// ---------------------------------------------------------------------------
__global__ __launch_bounds__(256) void xcast_kernel(const float* __restrict__ x,
                                                    __bf16* __restrict__ out) {
  const size_t base = ((size_t)blockIdx.x * 256 + threadIdx.x) * 8;
  #pragma unroll
  for (int j = 0; j < 8; ++j) out[base + j] = (__bf16)x[base + j];
}

// ---------------------------------------------------------------------------
// Persistent LSTM layer: one WG owns a 16-row batch tile for all T steps.
// Wih/Whh live in LDS as bf16 (CDNA5 320KB LDS). 8 waves x 4 N-tiles cover the
// 512 gate columns. The next step's x_t tile is double-buffered into LDS with
// GLOBAL_LOAD_ASYNC_TO_LDS_B128 while the current step's WMMAs run.
// ---------------------------------------------------------------------------
__global__ __launch_bounds__(256) void lstm_layer_kernel(
    const __bf16* __restrict__ xb,     // bf16 input [B,T,128]
    const float*  __restrict__ Wih,    // [512,128]
    const float*  __restrict__ Whh,    // [512,128]
    const float*  __restrict__ bih,    // [512]
    const float*  __restrict__ bhh,    // [512]
    __bf16* __restrict__ y_out,        // [B,T,128] bf16 or null
    float*  __restrict__ ctx_out,      // [B,128] mean_t(h) or null
    float*  __restrict__ h_last,       // [B,128] or null
    float*  __restrict__ c_last)       // [B,128] or null
{
  __shared__ __bf16 sWih[GATES * HID];     // 128 KB
  __shared__ __bf16 sWhh[GATES * HID];     // 128 KB
  __shared__ __bf16 sXdb[2][16 * 128];     // 8 KB (double-buffered x_t tile)
  __shared__ __bf16 sH[16 * 128];          // 4 KB
  __shared__ float  sGates[16 * GATES];    // 32 KB
  __shared__ float  sC[16 * 128];          // 8 KB
  __shared__ float  sCtx[16 * 128];        // 8 KB
  __shared__ float  sBias[GATES];          // 2 KB

  const int tid  = threadIdx.x;
  const int wave = tid >> 5;
  const int lane = tid & 31;
  const int row0 = blockIdx.x * 16;

  for (int i = tid; i < GATES * HID; i += 256) {
    sWih[i] = (__bf16)Wih[i];
    sWhh[i] = (__bf16)Whh[i];
  }
  for (int i = tid; i < GATES; i += 256) sBias[i] = bih[i] + bhh[i];
  for (int i = tid; i < 16 * 128; i += 256) {
    sH[i]   = (__bf16)0.0f;
    sC[i]   = 0.0f;
    sCtx[i] = 0.0f;
  }

  // Thread-static slice of the 16x128 tile: 8 contiguous bf16 = 16 bytes.
  const int pr = tid >> 4;           // tile row 0..15
  const int pf = (tid & 15) * 8;     // tile col 0..120

#if HAVE_ASYNC
  // Prologue: kick off tile t=0 into buffer 0.
  async_copy_b128(xb + ((size_t)(row0 + pr) * SEQT + 0) * 128 + pf,
                  &sXdb[0][tid * 8]);
#endif
  __syncthreads();

  for (int t = 0; t < SEQT; ++t) {
    const int cur = t & 1;
#if HAVE_ASYNC
    // Prefetch next tile into the other buffer (last iter: harmless re-load).
    {
      const int tn = (t + 1 < SEQT) ? (t + 1) : t;
      async_copy_b128(xb + ((size_t)(row0 + pr) * SEQT + tn) * 128 + pf,
                      &sXdb[cur ^ 1][tid * 8]);
    }
    wait_async_le1();   // this wave's tile-t portion landed; t+1 still in flight
#else
    for (int idx = tid; idx < 16 * 128; idx += 256) {
      const int r = idx >> 7, f = idx & 127;
      sXdb[cur][idx] = xb[((size_t)(row0 + r) * SEQT + t) * 128 + f];
    }
#endif
    __syncthreads();    // all waves' portions of tile t visible

    // gates = x_t @ Wih.T + h @ Whh.T  (f32 accumulate)
    v8f acc[4];
    #pragma unroll
    for (int q = 0; q < 4; ++q) acc[q] = zero8();

    #pragma unroll
    for (int kk = 0; kk < 4; ++kk) {
      const v16bf a = load_A_frag(&sXdb[cur][0], lane, kk);
      #pragma unroll
      for (int q = 0; q < 4; ++q) {
        const v16bf b = load_B_frag_lds(sWih, lane, (wave * 4 + q) * 16, kk, 128);
        acc[q] = WMMA_BF16(a, b, acc[q]);
      }
    }
    #pragma unroll
    for (int kk = 0; kk < 4; ++kk) {
      const v16bf a = load_A_frag(sH, lane, kk);
      #pragma unroll
      for (int q = 0; q < 4; ++q) {
        const v16bf b = load_B_frag_lds(sWhh, lane, (wave * 4 + q) * 16, kk, 128);
        acc[q] = WMMA_BF16(a, b, acc[q]);
      }
    }
    #pragma unroll
    for (int q = 0; q < 4; ++q)
      store_D(sGates, lane, (wave * 4 + q) * 16, GATES, acc[q]);
    __syncthreads();

    // Elementwise LSTM cell (PyTorch gate order i,f,g,o)
    for (int idx = tid; idx < 16 * 128; idx += 256) {
      const int r = idx >> 7, j = idx & 127;
      float ig = sGates[r * GATES + j]       + sBias[j];
      float fg = sGates[r * GATES + 128 + j] + sBias[128 + j];
      float gg = sGates[r * GATES + 256 + j] + sBias[256 + j];
      float og = sGates[r * GATES + 384 + j] + sBias[384 + j];
      ig = sigmoidf(ig);
      fg = sigmoidf(fg);
      gg = tanhf(gg);
      og = sigmoidf(og);
      const float c = fg * sC[idx] + ig * gg;
      const float h = og * tanhf(c);
      sC[idx]   = c;
      sH[idx]   = (__bf16)h;
      sCtx[idx] += h;
      if (y_out)
        y_out[((size_t)(row0 + r) * SEQT + t) * 128 + j] = (__bf16)h;
      if (t == SEQT - 1) {
        if (h_last) h_last[(row0 + r) * 128 + j] = h;
        if (c_last) c_last[(row0 + r) * 128 + j] = c;
      }
    }
    __syncthreads();
  }

  if (ctx_out) {
    const float inv = 1.0f / (float)SEQT;
    for (int idx = tid; idx < 16 * 128; idx += 256) {
      const int r = idx >> 7, j = idx & 127;
      ctx_out[(row0 + r) * 128 + j] = sCtx[idx] * inv;
    }
  }
}

// ---------------------------------------------------------------------------
// Mean over T of x: out[b,f] = mean_t x[b,t,f]
// ---------------------------------------------------------------------------
__global__ __launch_bounds__(256) void xmean_kernel(const float* __restrict__ x,
                                                    float* __restrict__ out) {
  const int idx = blockIdx.x * 256 + threadIdx.x;     // over B*F
  const int b = idx >> 7, f = idx & 127;
  const float* p = x + (size_t)b * SEQT * 128 + f;
  float s = 0.0f;
  for (int t = 0; t < SEQT; ++t) s += p[(size_t)t * 128];
  out[idx] = s * (1.0f / (float)SEQT);
}

// ---------------------------------------------------------------------------
// dec_in = xmean @ inproj_W.T + inproj_b    [1024,128] @ [128,128]
// ---------------------------------------------------------------------------
__global__ __launch_bounds__(256) void inproj_kernel(
    const float* __restrict__ xmean, const float* __restrict__ W,
    const float* __restrict__ bias, float* __restrict__ out) {
  __shared__ __bf16 sA[16 * 128];
  const int tid = threadIdx.x, wave = tid >> 5, lane = tid & 31;
  const int row0 = blockIdx.x * 16;

  for (int idx = tid; idx < 16 * 128; idx += 256) {
    const int r = idx >> 7, f = idx & 127;
    sA[idx] = (__bf16)xmean[(row0 + r) * 128 + f];
  }
  __syncthreads();

  v8f acc = zero8();
  #pragma unroll
  for (int kk = 0; kk < 4; ++kk) {
    const v16bf a = load_A_frag(sA, lane, kk);
    const v16bf b = load_B_frag_gf32(W, lane, wave * 16, kk, 128);
    acc = WMMA_BF16(a, b, acc);
  }
  const int n = wave * 16 + (lane & 15);
  const int half = lane >> 4;
  #pragma unroll
  for (int r = 0; r < 8; ++r)
    out[(row0 + r + 8 * half) * 128 + n] = acc[r] + bias[n];
}

// ---------------------------------------------------------------------------
// Autoregressive decoder: 15 steps; LSTM cell + additive attention proj +
// output projection. din(step0)=dec_in, then din = previous out.
// ---------------------------------------------------------------------------
__global__ __launch_bounds__(256) void decoder_kernel(
    const float* __restrict__ Wih, const float* __restrict__ Whh,
    const float* __restrict__ bih, const float* __restrict__ bhh,
    const float* __restrict__ attn_W,    // [128,256]
    const float* __restrict__ attn_b,    // [128]
    const float* __restrict__ outproj_W, // [128,128]
    const float* __restrict__ outproj_b, // [128]
    const float* __restrict__ ctx_in,    // [B,128]
    const float* __restrict__ h_in, const float* __restrict__ c_in,
    const float* __restrict__ din_in,    // [B,128]
    float* __restrict__ out)             // [B,15,128]
{
  __shared__ __bf16 sWih[GATES * HID];   // 128 KB
  __shared__ __bf16 sWhh[GATES * HID];   // 128 KB
  __shared__ __bf16 sDin[16 * 128];      // 4 KB
  __shared__ __bf16 sH[16 * 128];        // 4 KB
  __shared__ __bf16 sCtx[16 * 128];      // 4 KB
  __shared__ float  sGates[16 * GATES];  // 32 KB (reused as [16][128] scratch)
  __shared__ float  sC[16 * 128];        // 8 KB
  __shared__ float  sHf[16 * 128];       // 8 KB
  __shared__ float  sBias[GATES];        // 2 KB

  const int tid = threadIdx.x, wave = tid >> 5, lane = tid & 31;
  const int row0 = blockIdx.x * 16;

  for (int i = tid; i < GATES * HID; i += 256) {
    sWih[i] = (__bf16)Wih[i];
    sWhh[i] = (__bf16)Whh[i];
  }
  for (int i = tid; i < GATES; i += 256) sBias[i] = bih[i] + bhh[i];
  for (int idx = tid; idx < 16 * 128; idx += 256) {
    const int r = idx >> 7, j = idx & 127;
    const int g = (row0 + r) * 128 + j;
    sH[idx]   = (__bf16)h_in[g];
    sC[idx]   = c_in[g];
    sCtx[idx] = (__bf16)ctx_in[g];
    sDin[idx] = (__bf16)din_in[g];
  }
  __syncthreads();

  for (int step = 0; step < OUTLEN; ++step) {
    // gates = din @ Wih.T + h @ Whh.T
    v8f acc[4];
    #pragma unroll
    for (int q = 0; q < 4; ++q) acc[q] = zero8();

    #pragma unroll
    for (int kk = 0; kk < 4; ++kk) {
      const v16bf a = load_A_frag(sDin, lane, kk);
      #pragma unroll
      for (int q = 0; q < 4; ++q) {
        const v16bf b = load_B_frag_lds(sWih, lane, (wave * 4 + q) * 16, kk, 128);
        acc[q] = WMMA_BF16(a, b, acc[q]);
      }
    }
    #pragma unroll
    for (int kk = 0; kk < 4; ++kk) {
      const v16bf a = load_A_frag(sH, lane, kk);
      #pragma unroll
      for (int q = 0; q < 4; ++q) {
        const v16bf b = load_B_frag_lds(sWhh, lane, (wave * 4 + q) * 16, kk, 128);
        acc[q] = WMMA_BF16(a, b, acc[q]);
      }
    }
    #pragma unroll
    for (int q = 0; q < 4; ++q)
      store_D(sGates, lane, (wave * 4 + q) * 16, GATES, acc[q]);
    __syncthreads();

    // LSTM cell update
    for (int idx = tid; idx < 16 * 128; idx += 256) {
      const int r = idx >> 7, j = idx & 127;
      float ig = sGates[r * GATES + j]       + sBias[j];
      float fg = sGates[r * GATES + 128 + j] + sBias[128 + j];
      float gg = sGates[r * GATES + 256 + j] + sBias[256 + j];
      float og = sGates[r * GATES + 384 + j] + sBias[384 + j];
      ig = sigmoidf(ig);
      fg = sigmoidf(fg);
      gg = tanhf(gg);
      og = sigmoidf(og);
      const float c = fg * sC[idx] + ig * gg;
      const float h = og * tanhf(c);
      sC[idx]  = c;
      sHf[idx] = h;
      sH[idx]  = (__bf16)h;
    }
    __syncthreads();

    // attn: [16,256] = concat(h, ctx); result = concat @ attn_W.T -> [16,128]
    {
      v8f a2 = zero8();
      #pragma unroll
      for (int kk = 0; kk < 8; ++kk) {
        const v16bf a = (kk < 4) ? load_A_frag(sH, lane, kk)
                                 : load_A_frag(sCtx, lane, kk - 4);
        const v16bf b = load_B_frag_gf32(attn_W, lane, wave * 16, kk, 256);
        a2 = WMMA_BF16(a, b, a2);
      }
      store_D(sGates, lane, wave * 16, 128, a2);   // reuse as [16][128]
    }
    __syncthreads();

    for (int idx = tid; idx < 16 * 128; idx += 256) {
      const int j = idx & 127;
      const float h2 = sHf[idx] + sGates[idx] + attn_b[j];
      sHf[idx] = h2;
      sH[idx]  = (__bf16)h2;
    }
    __syncthreads();

    // out = h @ outproj_W.T + outproj_b -> [16,128]
    {
      v8f a3 = zero8();
      #pragma unroll
      for (int kk = 0; kk < 4; ++kk) {
        const v16bf a = load_A_frag(sH, lane, kk);
        const v16bf b = load_B_frag_gf32(outproj_W, lane, wave * 16, kk, 128);
        a3 = WMMA_BF16(a, b, a3);
      }
      store_D(sGates, lane, wave * 16, 128, a3);
    }
    __syncthreads();

    for (int idx = tid; idx < 16 * 128; idx += 256) {
      const int r = idx >> 7, j = idx & 127;
      const float o = sGates[idx] + outproj_b[j];
      out[((size_t)(row0 + r) * OUTLEN + step) * 128 + j] = o;
      sDin[idx] = (__bf16)o;   // next step's decoder input
    }
    __syncthreads();
  }
}

// ---------------------------------------------------------------------------
extern "C" void kernel_launch(void* const* d_in, const int* in_sizes, int n_in,
                              void* d_out, int out_size, void* d_ws, size_t ws_size,
                              hipStream_t stream) {
  (void)in_sizes; (void)n_in; (void)out_size; (void)ws_size;

  const float* x        = (const float*)d_in[0];
  const float* eWih0    = (const float*)d_in[1];
  const float* eWhh0    = (const float*)d_in[2];
  const float* ebih0    = (const float*)d_in[3];
  const float* ebhh0    = (const float*)d_in[4];
  const float* eWih1    = (const float*)d_in[5];
  const float* eWhh1    = (const float*)d_in[6];
  const float* ebih1    = (const float*)d_in[7];
  const float* ebhh1    = (const float*)d_in[8];
  const float* dWih     = (const float*)d_in[9];
  const float* dWhh     = (const float*)d_in[10];
  const float* dbih     = (const float*)d_in[11];
  const float* dbhh     = (const float*)d_in[12];
  const float* inprojW  = (const float*)d_in[13];
  const float* inprojB  = (const float*)d_in[14];
  const float* outprojW = (const float*)d_in[15];
  const float* outprojB = (const float*)d_in[16];
  const float* attnW    = (const float*)d_in[17];
  const float* attnB    = (const float*)d_in[18];

  // Workspace layout
  char* ws = (char*)d_ws;
  size_t off = 0;
  __bf16* xbf = (__bf16*)(ws + off); off += (size_t)BATCH * SEQT * FEAT * sizeof(__bf16);
  __bf16* y0  = (__bf16*)(ws + off); off += (size_t)BATCH * SEQT * HID * sizeof(__bf16);
  float* ctx   = (float*)(ws + off); off += (size_t)BATCH * HID * sizeof(float);
  float* hlast = (float*)(ws + off); off += (size_t)BATCH * HID * sizeof(float);
  float* clast = (float*)(ws + off); off += (size_t)BATCH * HID * sizeof(float);
  float* xmean = (float*)(ws + off); off += (size_t)BATCH * FEAT * sizeof(float);
  float* decin = (float*)(ws + off);

  const dim3 blk(256);

  // x f32 -> bf16 (one streaming pass)
  xcast_kernel<<<dim3((BATCH * SEQT * FEAT) / (256 * 8)), blk, 0, stream>>>(x, xbf);

  xmean_kernel<<<dim3(BATCH * FEAT / 256), blk, 0, stream>>>(x, xmean);

  // Encoder layer 0: bf16 x in, bf16 sequence out
  lstm_layer_kernel<<<dim3(BATCH / 16), blk, 0, stream>>>(
      xbf, eWih0, eWhh0, ebih0, ebhh0, y0,
      nullptr, nullptr, nullptr);

  // Encoder layer 1: bf16 seq in; fused ctx mean + final (h, c)
  lstm_layer_kernel<<<dim3(BATCH / 16), blk, 0, stream>>>(
      y0, eWih1, eWhh1, ebih1, ebhh1, nullptr,
      ctx, hlast, clast);

  inproj_kernel<<<dim3(BATCH / 16), blk, 0, stream>>>(xmean, inprojW, inprojB, decin);

  decoder_kernel<<<dim3(BATCH / 16), blk, 0, stream>>>(
      dWih, dWhh, dbih, dbhh, attnW, attnB, outprojW, outprojB,
      ctx, hlast, clast, decin, (float*)d_out);
}
